// Experts_66614942761445
// MI455X (gfx1250) — compile-verified, matrix operand order checked by reading, not run on hardware
//
#include <hip/hip_runtime.h>
#include <hip/hip_bf16.h>
#include <math.h>

// ---------------------------------------------------------------------------
// Problem constants (match reference)
// ---------------------------------------------------------------------------
#define NN 50000      // nodes
#define NE 800000     // edges
#define HD 64         // hidden = feature dim
#define KE 4          // experts
#define NL 3          // GIN layers
#define NB 256        // graphs in batch
#define NC 10         // classes

typedef __attribute__((ext_vector_type(16))) _Float16 v16h;
typedef __attribute__((ext_vector_type(8)))  _Float16 v8h;
typedef __attribute__((ext_vector_type(8)))  float    v8f;

// ---------------------------------------------------------------------------
// WMMA helpers (CDNA5: D = A(16x32 f16) * B(32x16 f16) + C(16x16 f32))
// ISA 7.12.2 layouts:
//  A: lane l holds row m=l&15; halves 0..7 -> K=(l>>4)*8+j, 8..15 -> K=16+(l>>4)*8+j
//  B: lane l holds col n=l&15; same K split by lane half
//  C/D: lane l col n=l&15; VGPR r -> row m=(l>>4)*8+r
// ---------------------------------------------------------------------------
static __device__ inline v8f wmma_f16(v16h a, v16h b, v8f c) {
  return __builtin_amdgcn_wmma_f32_16x16x32_f16(false, a, false, b, (short)0, c,
                                                false, false);
}

// A fragment from row-major f32 global, base offset to (row0, k0), row stride ld.
// All addresses 16B aligned (ld=64, k0 multiple of 32, kh in {0,8}).
static __device__ inline v16h loadA_f32v(const float* __restrict__ base, int ld, int lane) {
  const int m = lane & 15;
  const int kh = (lane >> 4) << 3;
  const float* r = base + m * ld + kh;
  const float4 p0 = *(const float4*)(r + 0);
  const float4 p1 = *(const float4*)(r + 4);
  const float4 p2 = *(const float4*)(r + 16);
  const float4 p3 = *(const float4*)(r + 20);
  v16h a;
  a[0] = (_Float16)p0.x;  a[1] = (_Float16)p0.y;  a[2] = (_Float16)p0.z;  a[3] = (_Float16)p0.w;
  a[4] = (_Float16)p1.x;  a[5] = (_Float16)p1.y;  a[6] = (_Float16)p1.z;  a[7] = (_Float16)p1.w;
  a[8] = (_Float16)p2.x;  a[9] = (_Float16)p2.y;  a[10] = (_Float16)p2.z; a[11] = (_Float16)p2.w;
  a[12] = (_Float16)p3.x; a[13] = (_Float16)p3.y; a[14] = (_Float16)p3.z; a[15] = (_Float16)p3.w;
  return a;
}

// Cooperative stage of a [kdim x 64] row-major f32 weight matrix into LDS in
// B-fragment-swizzled order: fragment for (tile t = ktIdx*4 + ct/16, lane) is
// 16 contiguous halves => B-fragment load is one 32B LDS read.
static __device__ inline void stageB(const float* __restrict__ W,
                                     _Float16* __restrict__ dstS, int kdim, int tid) {
  const int ntl = (kdim >> 5) * 4 * 32;  // tiles * 32 lanes
  for (int t16 = tid; t16 < ntl; t16 += 256) {
    const int t = t16 >> 5, lane = t16 & 31;
    const int kt = (t >> 2) << 5, ct = (t & 3) << 4;
    const int kh = (lane >> 4) << 3, n = lane & 15;
    const float* s0 = W + (kt + kh) * 64 + ct + n;
    const float* s1 = W + (kt + 16 + kh) * 64 + ct + n;
    v16h tmp;
#pragma unroll
    for (int j = 0; j < 8; ++j) tmp[j] = (_Float16)s0[j * 64];
#pragma unroll
    for (int j = 0; j < 8; ++j) tmp[8 + j] = (_Float16)s1[j * 64];
    *(v16h*)(dstS + (size_t)(t * 32 + lane) * 16) = tmp;
  }
}

static __device__ inline v16h loadBsw(const _Float16* __restrict__ sws, int tile, int lane) {
  return *(const v16h*)(sws + (size_t)(tile * 32 + lane) * 16);
}

// k-index permutation for A-layout hidden staging. Storage group order:
// [k0-7, k16-23, k8-15, k24-31, k32-39, k48-55, k40-47, k56-63] so that each
// (lane-half, kt-tile) reads 16 contiguous halves.
static __device__ inline int kperm(int kk) {
  const int b8 = kk >> 3;
  const int sg = (b8 & 4) | ((b8 & 1) << 1) | ((b8 >> 1) & 1);
  return sg * 8 + (kk & 7);
}

// A fragment from permuted f16 LDS staging (row stride 72 halves)
static __device__ inline v16h loadA_perm(const _Float16* __restrict__ rowbase, int ti, int lane) {
  const int m = lane & 15;
  const int hh = lane >> 4;
  const _Float16* p = rowbase + m * 72 + ti * 32 + hh * 16;
  const v8h lo = *(const v8h*)(p);
  const v8h hi = *(const v8h*)(p + 8);
  v16h a;
#pragma unroll
  for (int j = 0; j < 8; ++j) { a[j] = lo[j]; a[8 + j] = hi[j]; }
  return a;
}

static __device__ inline float sigmoidf_(float x) { return 1.0f / (1.0f + expf(-x)); }

// ---------------------------------------------------------------------------
// t[i] = (1 + eps[layer]) * h[i]
// ---------------------------------------------------------------------------
__global__ void k_scale_init(const float* __restrict__ h, float* __restrict__ t,
                             const float* __restrict__ eps, int layer, long count) {
  size_t i = (size_t)blockIdx.x * blockDim.x + threadIdx.x;
  if (i < (size_t)count) t[i] = (1.0f + eps[layer]) * h[i];
}

// ---------------------------------------------------------------------------
// Edge scatter: t[k][dst[e]][:] += (ew ? ew[k][e] : 1) * h[k][src[e]][:]
// one thread = one edge x 4 features; grid.y = expert
// ---------------------------------------------------------------------------
__global__ void k_scatter(const float* __restrict__ h, const int* __restrict__ src,
                          const int* __restrict__ dst, const float* __restrict__ ew,
                          float* __restrict__ t) {
  int tid = blockIdx.x * blockDim.x + threadIdx.x;
  int e = tid >> 4;
  if (e >= NE) return;
  int f4 = (tid & 15) << 2;
  int k = blockIdx.y;
  const float* hk = h + (size_t)k * NN * HD;
  float* tk = t + (size_t)k * NN * HD;
  int s = src[e], d = dst[e];
  float w = ew ? ew[(size_t)k * NE + e] : 1.0f;
  const float4 v = *(const float4*)(hk + (size_t)s * HD + f4);
  float* dp = tk + (size_t)d * HD + f4;
  atomicAdd(dp + 0, v.x * w);
  atomicAdd(dp + 1, v.y * w);
  atomicAdd(dp + 2, v.z * w);
  atomicAdd(dp + 3, v.w * w);
}

// ---------------------------------------------------------------------------
// Fused GIN MLP: hout = relu(relu(t @ W1 + b1) @ W2 + b2), rows x 64
// 8 waves/block, one 16-row tile per wave; swizzled weights + hid staging.
// ---------------------------------------------------------------------------
__global__ __launch_bounds__(256) void k_gin_mlp(
    const float* __restrict__ t, const float* __restrict__ W1,
    const float* __restrict__ b1, const float* __restrict__ W2,
    const float* __restrict__ b2, float* __restrict__ hout, int rows) {
  __shared__ __align__(64) _Float16 sW1[8 * 32 * 16];
  __shared__ __align__(64) _Float16 sW2[8 * 32 * 16];
  __shared__ __align__(16) float sb1[HD], sb2[HD];
  __shared__ __align__(64) _Float16 sHid[8][16 * 72];

  stageB(W1, sW1, HD, threadIdx.x);
  stageB(W2, sW2, HD, threadIdx.x);
  if (threadIdx.x < HD) {
    sb1[threadIdx.x] = b1[threadIdx.x];
    sb2[threadIdx.x] = b2[threadIdx.x];
  }
  __syncthreads();

  const int wave = threadIdx.x >> 5;
  const int lane = threadIdx.x & 31;
  const int tile = blockIdx.x * 8 + wave;
  if (tile * 16 >= rows) return;

  const float* A0 = t + (size_t)tile * 16 * HD;
  const int rb = (lane >> 4) << 3;
  const int nn = lane & 15;
  const v8f vz = {0.f, 0.f, 0.f, 0.f, 0.f, 0.f, 0.f, 0.f};

  v8f acc[4];
#pragma unroll
  for (int c = 0; c < 4; ++c) acc[c] = vz;
#pragma unroll
  for (int ti = 0; ti < 2; ++ti) {
    v16h a = loadA_f32v(A0 + ti * 32, HD, lane);
#pragma unroll
    for (int c = 0; c < 4; ++c)
      acc[c] = wmma_f16(a, loadBsw(sW1, ti * 4 + c, lane), acc[c]);
  }
#pragma unroll
  for (int c = 0; c < 4; ++c)
#pragma unroll
    for (int r = 0; r < 8; ++r) {
      float v = acc[c][r] + sb1[c * 16 + nn];
      sHid[wave][(rb + r) * 72 + kperm(c * 16 + nn)] = (_Float16)fmaxf(v, 0.f);
    }

  v8f acc2[4];
#pragma unroll
  for (int c = 0; c < 4; ++c) acc2[c] = vz;
#pragma unroll
  for (int ti = 0; ti < 2; ++ti) {
    v16h a = loadA_perm(sHid[wave], ti, lane);
#pragma unroll
    for (int c = 0; c < 4; ++c)
      acc2[c] = wmma_f16(a, loadBsw(sW2, ti * 4 + c, lane), acc2[c]);
  }
  float* O = hout + (size_t)tile * 16 * HD;
#pragma unroll
  for (int c = 0; c < 4; ++c)
#pragma unroll
    for (int r = 0; r < 8; ++r) {
      float v = acc2[c][r] + sb2[c * 16 + nn];
      O[(rb + r) * HD + c * 16 + nn] = fmaxf(v, 0.f);
    }
}

// ---------------------------------------------------------------------------
// Node mask: sigmoid(relu(Z @ nm_W1[k] + b1) @ nm_W2[k] + b2); grid.y = k
// ---------------------------------------------------------------------------
__global__ __launch_bounds__(256) void k_node_mask(
    const float* __restrict__ Z, const float* __restrict__ W1,
    const float* __restrict__ b1, const float* __restrict__ W2,
    const float* __restrict__ b2, float* __restrict__ nmv,
    float* __restrict__ out_nm) {
  const int k = blockIdx.y;
  __shared__ __align__(64) _Float16 sW1[8 * 32 * 16];
  __shared__ __align__(16) float sb1[HD], sw2[HD];
  __shared__ __align__(16) float sHidF[8][16 * 68];

  stageB(W1 + (size_t)k * HD * HD, sW1, HD, threadIdx.x);
  if (threadIdx.x < HD) {
    sb1[threadIdx.x] = b1[(size_t)k * HD + threadIdx.x];
    sw2[threadIdx.x] = W2[(size_t)k * HD + threadIdx.x];
  }
  __syncthreads();

  const int wave = threadIdx.x >> 5;
  const int lane = threadIdx.x & 31;
  const int tile = blockIdx.x * 8 + wave;
  if (tile * 16 >= NN) return;

  const float* A0 = Z + (size_t)tile * 16 * HD;
  const int rb = (lane >> 4) << 3;
  const int nn = lane & 15;
  const v8f vz = {0.f, 0.f, 0.f, 0.f, 0.f, 0.f, 0.f, 0.f};
  v8f acc[4];
#pragma unroll
  for (int c = 0; c < 4; ++c) acc[c] = vz;
#pragma unroll
  for (int ti = 0; ti < 2; ++ti) {
    v16h a = loadA_f32v(A0 + ti * 32, HD, lane);
#pragma unroll
    for (int c = 0; c < 4; ++c)
      acc[c] = wmma_f16(a, loadBsw(sW1, ti * 4 + c, lane), acc[c]);
  }
#pragma unroll
  for (int c = 0; c < 4; ++c)
#pragma unroll
    for (int r = 0; r < 8; ++r)
      sHidF[wave][(rb + r) * 68 + c * 16 + nn] = fmaxf(acc[c][r] + sb1[c * 16 + nn], 0.f);

  if (lane < 16) {
    float s = b2[k];
    const float4* hr = (const float4*)(sHidF[wave] + lane * 68);
    const float4* w4 = (const float4*)sw2;
#pragma unroll
    for (int j = 0; j < 16; ++j) {
      float4 hv = hr[j], wv = w4[j];
      s += hv.x * wv.x + hv.y * wv.y + hv.z * wv.z + hv.w * wv.w;
    }
    float sg = sigmoidf_(s);
    int node = tile * 16 + lane;
    nmv[(size_t)k * NN + node] = sg;
    out_nm[(size_t)node * KE + k] = sg;
  }
}

// ---------------------------------------------------------------------------
// Edge mask: A rows gathered on the fly from [Z[src]|Z[dst]] (128-wide); grid.y = k
// ---------------------------------------------------------------------------
__global__ __launch_bounds__(256) void k_edge_mask(
    const float* __restrict__ Z, const int* __restrict__ src,
    const int* __restrict__ dst, const float* __restrict__ W1,
    const float* __restrict__ b1, const float* __restrict__ W2,
    const float* __restrict__ b2, float* __restrict__ ewv,
    float* __restrict__ out_em) {
  const int k = blockIdx.y;
  __shared__ __align__(64) _Float16 sW1[16 * 32 * 16];  // 128 x 64 swizzled
  __shared__ __align__(16) float sb1[HD], sw2[HD];
  __shared__ __align__(16) float sHidF[8][16 * 68];

  stageB(W1 + (size_t)k * 2 * HD * HD, sW1, 2 * HD, threadIdx.x);
  if (threadIdx.x < HD) {
    sb1[threadIdx.x] = b1[(size_t)k * HD + threadIdx.x];
    sw2[threadIdx.x] = W2[(size_t)k * HD + threadIdx.x];
  }
  __syncthreads();

  const int wave = threadIdx.x >> 5;
  const int lane = threadIdx.x & 31;
  const int tile = blockIdx.x * 8 + wave;  // NE/16 = 50000 tiles exactly
  if (tile * 16 >= NE) return;

  const int m = lane & 15;
  const int e = tile * 16 + m;
  const int sn = src[e], dn = dst[e];
  const int rb = (lane >> 4) << 3;
  const int nn = lane & 15;
  const v8f vz = {0.f, 0.f, 0.f, 0.f, 0.f, 0.f, 0.f, 0.f};
  v8f acc[4];
#pragma unroll
  for (int c = 0; c < 4; ++c) acc[c] = vz;

#pragma unroll
  for (int ti = 0; ti < 4; ++ti) {
    // ti 0,1 -> Z[src], ti 2,3 -> Z[dst]; row-local k offset (ti&1)*32
    const float* zrow = Z + (size_t)((ti < 2) ? sn : dn) * HD + (ti & 1) * 32;
    const int kh = (lane >> 4) << 3;
    const float* r = zrow + kh;  // A row for this lane (m encoded via zrow per lane)
    const float4 p0 = *(const float4*)(r + 0);
    const float4 p1 = *(const float4*)(r + 4);
    const float4 p2 = *(const float4*)(r + 16);
    const float4 p3 = *(const float4*)(r + 20);
    v16h a;
    a[0] = (_Float16)p0.x;  a[1] = (_Float16)p0.y;  a[2] = (_Float16)p0.z;  a[3] = (_Float16)p0.w;
    a[4] = (_Float16)p1.x;  a[5] = (_Float16)p1.y;  a[6] = (_Float16)p1.z;  a[7] = (_Float16)p1.w;
    a[8] = (_Float16)p2.x;  a[9] = (_Float16)p2.y;  a[10] = (_Float16)p2.z; a[11] = (_Float16)p2.w;
    a[12] = (_Float16)p3.x; a[13] = (_Float16)p3.y; a[14] = (_Float16)p3.z; a[15] = (_Float16)p3.w;
#pragma unroll
    for (int c = 0; c < 4; ++c)
      acc[c] = wmma_f16(a, loadBsw(sW1, ti * 4 + c, lane), acc[c]);
  }
#pragma unroll
  for (int c = 0; c < 4; ++c)
#pragma unroll
    for (int r = 0; r < 8; ++r)
      sHidF[wave][(rb + r) * 68 + c * 16 + nn] = fmaxf(acc[c][r] + sb1[c * 16 + nn], 0.f);

  if (lane < 16) {
    float s = b2[k];
    const float4* hr = (const float4*)(sHidF[wave] + lane * 68);
    const float4* w4 = (const float4*)sw2;
#pragma unroll
    for (int j = 0; j < 16; ++j) {
      float4 hv = hr[j], wv = w4[j];
      s += hv.x * wv.x + hv.y * wv.y + hv.z * wv.z + hv.w * wv.w;
    }
    float sg = sigmoidf_(s);
    int e2 = tile * 16 + lane;
    ewv[(size_t)k * NE + e2] = sg;
    out_em[(size_t)e2 * KE + k] = sg;
  }
}

// ---------------------------------------------------------------------------
// Feature mask (2-stage WMMA) fused with masked_x = x * nm * fm; grid.y = k
// ---------------------------------------------------------------------------
__global__ __launch_bounds__(256) void k_feat_mask(
    const float* __restrict__ Z, const float* __restrict__ x,
    const float* __restrict__ W1, const float* __restrict__ b1,
    const float* __restrict__ W2, const float* __restrict__ b2,
    const float* __restrict__ nmv, float* __restrict__ out_fm,
    float* __restrict__ maskedx) {
  const int k = blockIdx.y;
  __shared__ __align__(64) _Float16 sW1[8 * 32 * 16];
  __shared__ __align__(64) _Float16 sW2[8 * 32 * 16];
  __shared__ __align__(16) float sb1[HD], sb2[HD];
  __shared__ __align__(64) _Float16 sHid[8][16 * 72];

  stageB(W1 + (size_t)k * HD * HD, sW1, HD, threadIdx.x);
  stageB(W2 + (size_t)k * HD * HD, sW2, HD, threadIdx.x);
  if (threadIdx.x < HD) {
    sb1[threadIdx.x] = b1[(size_t)k * HD + threadIdx.x];
    sb2[threadIdx.x] = b2[(size_t)k * HD + threadIdx.x];
  }
  __syncthreads();

  const int wave = threadIdx.x >> 5;
  const int lane = threadIdx.x & 31;
  const int tile = blockIdx.x * 8 + wave;
  if (tile * 16 >= NN) return;

  const float* A0 = Z + (size_t)tile * 16 * HD;
  const int rb = (lane >> 4) << 3;
  const int nn = lane & 15;
  const v8f vz = {0.f, 0.f, 0.f, 0.f, 0.f, 0.f, 0.f, 0.f};
  v8f acc[4];
#pragma unroll
  for (int c = 0; c < 4; ++c) acc[c] = vz;
#pragma unroll
  for (int ti = 0; ti < 2; ++ti) {
    v16h a = loadA_f32v(A0 + ti * 32, HD, lane);
#pragma unroll
    for (int c = 0; c < 4; ++c)
      acc[c] = wmma_f16(a, loadBsw(sW1, ti * 4 + c, lane), acc[c]);
  }
#pragma unroll
  for (int c = 0; c < 4; ++c)
#pragma unroll
    for (int r = 0; r < 8; ++r) {
      float v = acc[c][r] + sb1[c * 16 + nn];
      sHid[wave][(rb + r) * 72 + kperm(c * 16 + nn)] = (_Float16)fmaxf(v, 0.f);
    }

  v8f acc2[4];
#pragma unroll
  for (int c = 0; c < 4; ++c) acc2[c] = vz;
#pragma unroll
  for (int ti = 0; ti < 2; ++ti) {
    v16h a = loadA_perm(sHid[wave], ti, lane);
#pragma unroll
    for (int c = 0; c < 4; ++c)
      acc2[c] = wmma_f16(a, loadBsw(sW2, ti * 4 + c, lane), acc2[c]);
  }

#pragma unroll
  for (int c = 0; c < 4; ++c)
#pragma unroll
    for (int r = 0; r < 8; ++r) {
      int node = tile * 16 + rb + r;
      int f = c * 16 + nn;
      float sg = sigmoidf_(acc2[c][r] + sb2[f]);
      out_fm[(size_t)node * (KE * HD) + (size_t)k * HD + f] = sg;
      maskedx[((size_t)k * NN + node) * HD + f] =
          x[(size_t)node * HD + f] * nmv[(size_t)k * NN + node] * sg;
    }
}

// ---------------------------------------------------------------------------
// Pooling / finalize
// ---------------------------------------------------------------------------
__global__ void k_zero(float* __restrict__ p, long n) {
  size_t i = (size_t)blockIdx.x * blockDim.x + threadIdx.x;
  if (i < (size_t)n) p[i] = 0.f;
}

__global__ void k_count(const int* __restrict__ batch, float* __restrict__ cnt) {
  int i = blockIdx.x * blockDim.x + threadIdx.x;
  if (i < NN) atomicAdd(&cnt[batch[i]], 1.f);
}

__global__ void k_pool_sum(const float* __restrict__ z, const int* __restrict__ batch,
                           float* __restrict__ sum, long zStrideK, long sStrideK) {
  int k = blockIdx.y;
  int tid = blockIdx.x * blockDim.x + threadIdx.x;
  if (tid >= NN * HD) return;
  int n = tid >> 6, f = tid & 63;
  atomicAdd(&sum[(size_t)k * sStrideK + (size_t)batch[n] * HD + f],
            z[(size_t)k * zStrideK + tid]);
}

__global__ void k_finalize(const float* __restrict__ sumZ, const float* __restrict__ sumS,
                           const float* __restrict__ cnt, float* __restrict__ out_horig,
                           float* __restrict__ out_hstable) {
  int tid = blockIdx.x * blockDim.x + threadIdx.x;
  if (tid >= NB * HD) return;
  int b = tid >> 6, h = tid & 63;
  float c = fmaxf(cnt[b], 1.f);
  out_horig[tid] = sumZ[tid] / c;
#pragma unroll
  for (int k = 0; k < KE; ++k)
    out_hstable[(size_t)b * KE * HD + (size_t)k * HD + h] =
        sumS[((size_t)k * NB + b) * HD + h] / c;
}

__global__ void k_logits(const float* __restrict__ hst, const float* __restrict__ clfW,
                         const float* __restrict__ clfb, float* __restrict__ out) {
  int tid = blockIdx.x * blockDim.x + threadIdx.x;
  if (tid >= NB * KE * NC) return;
  int c = tid % NC;
  int k = (tid / NC) % KE;
  int b = tid / (NC * KE);
  float s = clfb[k * NC + c];
  const float* hr = hst + (size_t)b * KE * HD + (size_t)k * HD;
#pragma unroll
  for (int h = 0; h < HD; ++h) s += hr[h] * clfW[((size_t)k * HD + h) * NC + c];
  out[tid] = s;
}

// ---------------------------------------------------------------------------
// Launch: workspace needs ~185 MiB (48,282,176 floats)
// ---------------------------------------------------------------------------
extern "C" void kernel_launch(void* const* d_in, const int* in_sizes, int n_in,
                              void* d_out, int out_size, void* d_ws, size_t ws_size,
                              hipStream_t stream) {
  const float* x = (const float*)d_in[0];
  const int* ei = (const int*)d_in[1];
  const int* src = ei;
  const int* dst = ei + NE;
  const int* batch = (const int*)d_in[2];
  const float* ce_W1 = (const float*)d_in[3];
  const float* ce_b1 = (const float*)d_in[4];
  const float* ce_W2 = (const float*)d_in[5];
  const float* ce_b2 = (const float*)d_in[6];
  const float* ce_eps = (const float*)d_in[7];
  const float* cl_W1 = (const float*)d_in[8];
  const float* cl_b1 = (const float*)d_in[9];
  const float* cl_W2 = (const float*)d_in[10];
  const float* cl_b2 = (const float*)d_in[11];
  const float* cl_eps = (const float*)d_in[12];
  const float* nm_W1 = (const float*)d_in[13];
  const float* nm_b1 = (const float*)d_in[14];
  const float* nm_W2 = (const float*)d_in[15];
  const float* nm_b2 = (const float*)d_in[16];
  const float* em_W1 = (const float*)d_in[17];
  const float* em_b1 = (const float*)d_in[18];
  const float* em_W2 = (const float*)d_in[19];
  const float* em_b2 = (const float*)d_in[20];
  const float* fm_W1 = (const float*)d_in[21];
  const float* fm_b1 = (const float*)d_in[22];
  const float* fm_W2 = (const float*)d_in[23];
  const float* fm_b2 = (const float*)d_in[24];
  const float* clf_W = (const float*)d_in[25];
  const float* clf_b = (const float*)d_in[26];

  // output layout (flat concat, reference return order)
  float* out = (float*)d_out;
  float* out_logits = out;                            // [B,K,C]   10240
  float* out_hstable = out + 10240;                   // [B,K,H]   65536
  float* out_horig = out + 75776;                     // [B,H]     16384
  float* out_nm = out + 92160;                        // [N,K,1]   200000
  float* out_em = out + 292160;                       // [E,K,1]   3200000
  float* out_fm = out + 3492160;                      // [N,K,F]   12800000

  // workspace layout
  float* ws = (float*)d_ws;
  size_t o = 0;
  float* Zb0 = ws + o;  o += (size_t)NN * HD;
  float* Zb1 = ws + o;  o += (size_t)NN * HD;
  float* tbuf = ws + o; o += (size_t)KE * NN * HD;    // slice 0 for encoder
  float* hA = ws + o;   o += (size_t)KE * NN * HD;    // masked_x / ping
  float* hB = ws + o;   o += (size_t)KE * NN * HD;    // pong
  float* nmv = ws + o;  o += (size_t)KE * NN;         // node masks [K,N]
  float* ewv = ws + o;  o += (size_t)KE * NE;         // edge weights [K,E]
  float* sumZ = ws + o; o += (size_t)NB * HD;
  float* sumS = ws + o; o += (size_t)KE * NB * HD;
  float* cnt = ws + o;  o += (size_t)NB;

  dim3 blk(256);
  const int mlpBlocksN = (NN / 16 + 7) / 8;           // 391
  const int mlpBlocksKN = (KE * NN / 16 + 7) / 8;     // 1563
  const int scatterBlocks = (NE * 16) / 256;          // 50000
  const int initBlocksN = (NN * HD) / 256;            // 12500
  const int initBlocksKN = (KE * NN * HD) / 256;      // 50000

  // ---- causal encoder: 3 GIN layers on x -> Z (in Zb0) ----
  const float* hin = x;
  float* houts[NL] = {Zb0, Zb1, Zb0};
  for (int l = 0; l < NL; ++l) {
    k_scale_init<<<initBlocksN, blk, 0, stream>>>(hin, tbuf, ce_eps, l, (long)NN * HD);
    k_scatter<<<dim3(scatterBlocks, 1), blk, 0, stream>>>(hin, src, dst, nullptr, tbuf);
    k_gin_mlp<<<mlpBlocksN, blk, 0, stream>>>(tbuf, ce_W1 + l * HD * HD, ce_b1 + l * HD,
                                              ce_W2 + l * HD * HD, ce_b2 + l * HD,
                                              houts[l], NN);
    hin = houts[l];
  }
  const float* Z = Zb0;

  // ---- masks (grid.y = expert) ----
  k_node_mask<<<dim3(mlpBlocksN, KE), blk, 0, stream>>>(Z, nm_W1, nm_b1, nm_W2, nm_b2,
                                                        nmv, out_nm);
  k_feat_mask<<<dim3(mlpBlocksN, KE), blk, 0, stream>>>(Z, x, fm_W1, fm_b1, fm_W2, fm_b2,
                                                        nmv, out_fm, hA);
  k_edge_mask<<<dim3((NE / 16) / 8, KE), blk, 0, stream>>>(Z, src, dst, em_W1, em_b1,
                                                           em_W2, em_b2, ewv, out_em);

  // ---- expert GINs: 3 layers on masked_x with edge weights (shared cl weights) ----
  float* cur = hA;
  float* nxt = hB;
  for (int l = 0; l < NL; ++l) {
    k_scale_init<<<initBlocksKN, blk, 0, stream>>>(cur, tbuf, cl_eps, l,
                                                   (long)KE * NN * HD);
    k_scatter<<<dim3(scatterBlocks, KE), blk, 0, stream>>>(cur, src, dst, ewv, tbuf);
    k_gin_mlp<<<mlpBlocksKN, blk, 0, stream>>>(tbuf, cl_W1 + l * HD * HD, cl_b1 + l * HD,
                                               cl_W2 + l * HD * HD, cl_b2 + l * HD, nxt,
                                               KE * NN);
    float* tmp = cur; cur = nxt; nxt = tmp;
  }
  // after 3 swaps: cur == hB holds masked_Z [K,N,H]

  // ---- pooling + classifier ----
  long zeroN = (long)NB * HD + (long)KE * NB * HD + NB;  // sumZ,sumS,cnt contiguous
  k_zero<<<(int)((zeroN + 255) / 256), blk, 0, stream>>>(sumZ, zeroN);
  k_count<<<(NN + 255) / 256, blk, 0, stream>>>(batch, cnt);
  k_pool_sum<<<dim3(initBlocksN, 1), blk, 0, stream>>>(Z, batch, sumZ, 0, 0);
  k_pool_sum<<<dim3(initBlocksN, KE), blk, 0, stream>>>(cur, batch, sumS,
                                                        (long)NN * HD, (long)NB * HD);
  k_finalize<<<(NB * HD) / 256, blk, 0, stream>>>(sumZ, sumS, cnt, out_horig,
                                                  out_hstable);
  k_logits<<<(NB * KE * NC + 255) / 256, blk, 0, stream>>>(out_hstable, clf_W, clf_b,
                                                           out_logits);
}
